// MVDR_14886356648893
// MI455X (gfx1250) — matrix operand order repaired and checked by reading, not compiled, not run on hardware
//
#include <hip/hip_runtime.h>
#include <hip/hip_bf16.h>

// ---------------- problem constants ----------------
#define B_    8
#define L_    160000
#define C_    4
#define NFFT_ 320
#define HOP_  160
#define F_    161          // rfft bins
#define T_    1001         // frames
#define M1_   (B_*C_*T_)   // 32032 STFT gemm rows (mult of 16)
#define N1_   384          // 2*F=322 padded to 64 (6 supertiles)
#define K1_   320          // NFFT (mult of 32)
#define M2_   8016         // B*T=8008 padded to 16
#define K2_   352          // 2*F=322 padded to 32
#define N2_   320          // NFFT (5 supertiles of 64)
#define OUTLEN_ (B_*L_)    // per-beam output elements

typedef __attribute__((ext_vector_type(16))) _Float16 v16h;
typedef __attribute__((ext_vector_type(8)))  float    v8f;

// ---------------- complex helpers ----------------
struct cpx { float x, y; };
__device__ __forceinline__ cpx cadd(cpx a, cpx b){ return {a.x+b.x, a.y+b.y}; }
__device__ __forceinline__ cpx csub(cpx a, cpx b){ return {a.x-b.x, a.y-b.y}; }
__device__ __forceinline__ cpx cmul(cpx a, cpx b){ return {a.x*b.x - a.y*b.y, a.x*b.y + a.y*b.x}; }
__device__ __forceinline__ cpx cmulc(cpx a, cpx b){ // a * conj(b)
  return {a.x*b.x + a.y*b.y, a.y*b.x - a.x*b.y};
}
__device__ __forceinline__ cpx cinv(cpx a){
  float id = 1.0f / (a.x*a.x + a.y*a.y + 1e-30f);
  return {a.x*id, -a.y*id};
}
__device__ __forceinline__ cpx cdiv(cpx a, cpx b){ return cmul(a, cinv(b)); }

// ---------------- per-(b,c) mean over time ----------------
__global__ void mean_kernel(const float* __restrict__ wav, float* __restrict__ mean) {
  int bc = blockIdx.x;            // 0..31
  int b = bc >> 2, c = bc & 3;
  float s = 0.f;
  for (int l = threadIdx.x; l < L_; l += blockDim.x)
    s += wav[((size_t)b*L_ + l)*C_ + c];
  __shared__ float sm[256];
  sm[threadIdx.x] = s; __syncthreads();
  for (int off = 128; off; off >>= 1) {
    if (threadIdx.x < off) sm[threadIdx.x] += sm[threadIdx.x + off];
    __syncthreads();
  }
  if (threadIdx.x == 0) mean[bc] = sm[0] * (1.0f / (float)L_);
}

// ---------------- per-b 1/max|x-mean| ----------------
__global__ void maxabs_kernel(const float* __restrict__ wav, const float* __restrict__ mean,
                              float* __restrict__ invmax) {
  int b = blockIdx.x;             // 0..7
  float mx = 0.f;
  for (int i = threadIdx.x; i < L_*C_; i += blockDim.x) {
    int c = i & 3;
    float v = fabsf(wav[(size_t)b*L_*C_ + i] - mean[b*C_ + c]);
    mx = fmaxf(mx, v);
  }
  __shared__ float sm[256];
  sm[threadIdx.x] = mx; __syncthreads();
  for (int off = 128; off; off >>= 1) {
    if (threadIdx.x < off) sm[threadIdx.x] = fmaxf(sm[threadIdx.x], sm[threadIdx.x + off]);
    __syncthreads();
  }
  if (threadIdx.x == 0) invmax[b] = 1.0f / sm[0];
}

// ---------------- precompute DFT matrices (transposed) + steering vectors ----------------
// dftT : [N1][K1] f16; rows n<161 = win[k]*cos, 161..321 = -win[k]*sin, rest 0
// istT : [N2][K2] f16; cols k<161 = (s/320)*cos, 161..321 = -(s/320)*sin, rest 0
// h1,h2: [F][C] complex steering (angles -40 / +40 deg)
__global__ void precompute_kernel(const float* __restrict__ window,
                                  _Float16* __restrict__ dftT, _Float16* __restrict__ istT,
                                  float2* __restrict__ h1, float2* __restrict__ h2) {
  const float W = 6.2831853071795864769f / (float)NFFT_;   // 2*pi/320
  const int R1 = N1_*K1_, R2 = N2_*K2_, R3 = 2*F_*C_;
  for (int id = blockIdx.x*blockDim.x + threadIdx.x; id < R1 + R2 + R3;
       id += gridDim.x*blockDim.x) {
    if (id < R1) {
      int n = id / K1_, k = id % K1_;
      float v = 0.f;
      if (n < F_) {
        int m = (k * n) % NFFT_;                 // exact integer phase reduction
        v = window[k] * __cosf(W * (float)m);
      } else if (n < 2*F_) {
        int f = n - F_;
        int m = (k * f) % NFFT_;
        v = -window[k] * __sinf(W * (float)m);
      }
      dftT[id] = (_Float16)v;
    } else if (id < R1 + R2) {
      int id2 = id - R1;
      int n = id2 / K2_, k = id2 % K2_;
      float v = 0.f;
      if (k < F_) {
        float s = (k == 0 || k == F_-1) ? 1.f : 2.f;
        int m = (k * n) % NFFT_;
        v = (s / (float)NFFT_) * __cosf(W * (float)m);
      } else if (k < 2*F_) {
        int f = k - F_;
        float s = (f == 0 || f == F_-1) ? 1.f : 2.f;
        int m = (f * n) % NFFT_;
        v = -(s / (float)NFFT_) * __sinf(W * (float)m);
      }
      istT[id2] = (_Float16)v;
    } else {
      int id3 = id - R1 - R2;
      int which = id3 / (F_*C_);
      int r = id3 % (F_*C_);
      int f = r / C_, c = r % C_;
      const float SIN40 = 0.64278760968653932632f;
      float sa = which ? SIN40 : -SIN40;         // h1 = -40deg, h2 = +40deg
      // phi = 2*pi * (50*f) * (c*0.027*sin(theta)/340)
      float phi = 0.024946918f * (float)f * (float)c * sa;
      float2 hv; hv.x = __cosf(phi); hv.y = __sinf(phi);
      if (which) h2[r] = hv; else h1[r] = hv;
    }
  }
}

// ---------------- framing (reflect pad, mean/max normalize) -> f16 A matrix ----------------
__global__ void frame_kernel(const float* __restrict__ wav, const float* __restrict__ mean,
                             const float* __restrict__ invmax, _Float16* __restrict__ frames) {
  const int TOT = M1_ * K1_;
  for (int id = blockIdx.x*blockDim.x + threadIdx.x; id < TOT; id += gridDim.x*blockDim.x) {
    int k = id % K1_;
    int m = id / K1_;
    int t = m % T_;
    int bc = m / T_;
    int c = bc & 3, b = bc >> 2;
    int j = t*HOP_ + k - (NFFT_/2);
    j = (j < 0) ? -j : ((j >= L_) ? (2*L_ - 2 - j) : j);
    float v = (wav[((size_t)b*L_ + j)*C_ + c] - mean[bc]) * invmax[b];
    frames[id] = (_Float16)v;
  }
}

// ---------------- WMMA GEMM: C[M,N] = A[M,K] * Bt[N,K]^T ----------------
// One wave computes a 16x64 output tile: one A fragment reused across 4 B
// fragments / 4 accumulators per k-step, with double-buffered fragment loads
// so global_load_b128 overlaps v_wmma issue. Requires: M%16==0, N%64==0,
// K%32==0, all buffers 16B aligned.
__device__ __forceinline__ void ld_fragA(const _Float16* p, v16h& a) {
  reinterpret_cast<float4*>(&a)[0] = *reinterpret_cast<const float4*>(p);
  reinterpret_cast<float4*>(&a)[1] = *reinterpret_cast<const float4*>(p + 16);
}
__device__ __forceinline__ void ld_fragB(const _Float16* p, v16h& b) {
  reinterpret_cast<float4*>(&b)[0] = *reinterpret_cast<const float4*>(p);
  reinterpret_cast<float4*>(&b)[1] = *reinterpret_cast<const float4*>(p + 16);
}

__global__ void wmma_gemm_kernel(const _Float16* __restrict__ A,
                                 const _Float16* __restrict__ Bt,
                                 float* __restrict__ Cmat,
                                 int M, int N, int K) {
  int lane = threadIdx.x;                   // 0..31
  int wid  = blockIdx.x * blockDim.y + threadIdx.y;
  int tilesN = N >> 6;                      // 64-wide supertiles
  int tilesM = M >> 4;
  if (wid >= tilesM * tilesN) return;       // wave-uniform exit (EXEC stays all-ones)
  int tm = (wid / tilesN) << 4;
  int tn = (wid % tilesN) << 6;
  int hi = lane >> 4;                       // half-wave select
  int lo = lane & 15;

  // A 16x32 f16 layout: elems 0..7 -> K = k0+8*hi+e ; elems 8..15 -> K = k0+16+8*hi+e
  const _Float16* arow = A  + (size_t)(tm + lo) * K + 8 * hi;
  // B 32x16 f16 layout: elems 0..7 -> K = k0+8*hi+e ; elems 8..15 -> K = k0+16+8*hi+e
  const _Float16* bcol = Bt + (size_t)(tn + lo) * K + 8 * hi;
  const size_t bstep = (size_t)16 * K;      // next 16-column subtile

  v8f acc0 = {}, acc1 = {}, acc2 = {}, acc3 = {};
  v16h a, b0, b1, b2, b3;
  ld_fragA(arow, a);
  ld_fragB(bcol,             b0);
  ld_fragB(bcol + bstep,     b1);
  ld_fragB(bcol + 2*bstep,   b2);
  ld_fragB(bcol + 3*bstep,   b3);

  int k0 = 0;
  for (; k0 + 32 < K; k0 += 32) {
    v16h an, b0n, b1n, b2n, b3n;
    int kn = k0 + 32;
    ld_fragA(arow + kn, an);
    ld_fragB(bcol + kn,           b0n);
    ld_fragB(bcol + bstep + kn,   b1n);
    ld_fragB(bcol + 2*bstep + kn, b2n);
    ld_fragB(bcol + 3*bstep + kn, b3n);
    __builtin_prefetch(arow + kn + 32, 0, 0);
    acc0 = __builtin_amdgcn_wmma_f32_16x16x32_f16(false, a, false, b0, (short)0, acc0, false, false);
    acc1 = __builtin_amdgcn_wmma_f32_16x16x32_f16(false, a, false, b1, (short)0, acc1, false, false);
    acc2 = __builtin_amdgcn_wmma_f32_16x16x32_f16(false, a, false, b2, (short)0, acc2, false, false);
    acc3 = __builtin_amdgcn_wmma_f32_16x16x32_f16(false, a, false, b3, (short)0, acc3, false, false);
    a = an; b0 = b0n; b1 = b1n; b2 = b2n; b3 = b3n;
  }
  acc0 = __builtin_amdgcn_wmma_f32_16x16x32_f16(false, a, false, b0, (short)0, acc0, false, false);
  acc1 = __builtin_amdgcn_wmma_f32_16x16x32_f16(false, a, false, b1, (short)0, acc1, false, false);
  acc2 = __builtin_amdgcn_wmma_f32_16x16x32_f16(false, a, false, b2, (short)0, acc2, false, false);
  acc3 = __builtin_amdgcn_wmma_f32_16x16x32_f16(false, a, false, b3, (short)0, acc3, false, false);

  // D layout: VGPR r -> row tm + r + 8*hi, col tn + lo (+16 per subtile)
  #pragma unroll
  for (int r = 0; r < 8; ++r) {
    float* crow = Cmat + (size_t)(tm + r + 8*hi) * N + tn + lo;
    crow[0]  = acc0[r];
    crow[16] = acc1[r];
    crow[32] = acc2[r];
    crow[48] = acc3[r];
  }
}

// ---------------- zero-init padded f16 spectra for ISTFT GEMM ----------------
__global__ void zero_half_kernel(_Float16* __restrict__ p, int n) {
  for (int id = blockIdx.x*blockDim.x + threadIdx.x; id < n; id += gridDim.x*blockDim.x)
    p[id] = (_Float16)0.0f;
}

// ---------------- MVDR recursive scan: one lane per (b,f) ----------------
__global__ void mvdr_scan_kernel(const float* __restrict__ spec,
                                 const float2* __restrict__ h1g, const float2* __restrict__ h2g,
                                 _Float16* __restrict__ Sg1, _Float16* __restrict__ Sg2) {
  int tid = blockIdx.x*blockDim.x + threadIdx.x;
  if (tid >= B_*F_) return;
  int b = tid / F_, f = tid % F_;

  cpx h1[C_], h2[C_];
  #pragma unroll
  for (int c = 0; c < C_; ++c) {
    float2 v1 = h1g[f*C_ + c], v2 = h2g[f*C_ + c];
    h1[c] = {v1.x, v1.y}; h2[c] = {v2.x, v2.y};
  }

  cpx R[C_][C_];
  #pragma unroll
  for (int i = 0; i < C_; ++i)
    #pragma unroll
    for (int j = 0; j < C_; ++j) R[i][j] = {0.f, 0.f};

  for (int t = 0; t < T_; ++t) {
    // gather X[b, c, t, f] from spec rows ((b*4+c)*T + t)
    cpx X[C_];
    #pragma unroll
    for (int c = 0; c < C_; ++c) {
      size_t row = ((size_t)(b*C_ + c) * T_ + t) * N1_;
      X[c] = {spec[row + f], spec[row + F_ + f]};
    }
    // recursive covariance
    #pragma unroll
    for (int i = 0; i < C_; ++i)
      #pragma unroll
      for (int j = 0; j < C_; ++j) {
        cpx o = cmulc(X[i], X[j]);                // X_i * conj(X_j)
        if (t == 0) R[i][j] = o;
        else        R[i][j] = {0.05f*R[i][j].x + 0.95f*o.x,
                               0.05f*R[i][j].y + 0.95f*o.y};
      }
    // diagonal loading
    float dl = 0.25f * (R[0][0].x + R[1][1].x + R[2][2].x + R[3][3].x);
    cpx A[C_][C_];
    #pragma unroll
    for (int i = 0; i < C_; ++i)
      #pragma unroll
      for (int j = 0; j < C_; ++j) {
        A[i][j] = R[i][j];
        if (i == j) A[i][j].x += dl;
      }
    // Gauss-Jordan solve A y = h (2 RHS); diag-loaded Hermitian -> no pivoting
    cpx y1[C_], y2[C_];
    #pragma unroll
    for (int c = 0; c < C_; ++c) { y1[c] = h1[c]; y2[c] = h2[c]; }
    #pragma unroll
    for (int k = 0; k < C_; ++k) {
      cpx inv = cinv(A[k][k]);
      #pragma unroll
      for (int j = 0; j < C_; ++j) if (j > k) A[k][j] = cmul(A[k][j], inv);
      y1[k] = cmul(y1[k], inv);
      y2[k] = cmul(y2[k], inv);
      #pragma unroll
      for (int i = 0; i < C_; ++i) {
        if (i == k) continue;
        cpx fac = A[i][k];
        #pragma unroll
        for (int j = 0; j < C_; ++j) if (j > k) A[i][j] = csub(A[i][j], cmul(fac, A[k][j]));
        y1[i] = csub(y1[i], cmul(fac, y1[k]));
        y2[i] = csub(y2[i], cmul(fac, y2[k]));
      }
    }
    // den = h^H y, num = y^H X, out = num / conj(den)
    cpx d1 = {0,0}, d2 = {0,0}, n1 = {0,0}, n2 = {0,0};
    #pragma unroll
    for (int c = 0; c < C_; ++c) {
      d1 = cadd(d1, cmulc(y1[c], h1[c]));
      d2 = cadd(d2, cmulc(y2[c], h2[c]));
      n1 = cadd(n1, cmulc(X[c], y1[c]));
      n2 = cadd(n2, cmulc(X[c], y2[c]));
    }
    cpx o1 = cdiv(n1, {d1.x, -d1.y});
    cpx o2 = cdiv(n2, {d2.x, -d2.y});
    size_t rr = (size_t)(b*T_ + t) * K2_;
    Sg1[rr + f]       = (_Float16)o1.x;
    Sg1[rr + F_ + f]  = (_Float16)o1.y;
    Sg2[rr + f]       = (_Float16)o2.x;
    Sg2[rr + F_ + f]  = (_Float16)o2.y;
  }
}

// ---------------- overlap-add + coverage normalize + unpad ----------------
__global__ void ola_kernel(const float* __restrict__ framesO, float* __restrict__ out) {
  for (int id = blockIdx.x*blockDim.x + threadIdx.x; id < OUTLEN_;
       id += gridDim.x*blockDim.x) {
    int l = id % L_;
    int b = id / L_;
    int j = l + NFFT_/2;                  // padded coordinate
    int t0 = j / HOP_;
    float acc = 0.f; float cnt = 0.f;
    #pragma unroll
    for (int d = -1; d <= 0; ++d) {
      int t = t0 + d;
      if (t >= 0 && t < T_) {
        int off = j - t*HOP_;
        if (off >= 0 && off < NFFT_) {
          acc += framesO[((size_t)(b*T_ + t))*N2_ + off];
          cnt += 1.f;
        }
      }
    }
    out[id] = acc / cnt;
  }
}

// ---------------- launch ----------------
extern "C" void kernel_launch(void* const* d_in, const int* in_sizes, int n_in,
                              void* d_out, int out_size, void* d_ws, size_t ws_size,
                              hipStream_t stream) {
  (void)in_sizes; (void)n_in; (void)out_size; (void)ws_size;
  const float* wav    = (const float*)d_in[0];   // [B, L, C]
  const float* window = (const float*)d_in[1];   // [320]
  float* out = (float*)d_out;                    // two beams concatenated

  // workspace layout (256B aligned slices)
  char* ws = (char*)d_ws;
  size_t off = 0;
  auto take = [&](size_t bytes) -> char* {
    char* p = ws + off;
    off = (off + bytes + 255) & ~(size_t)255;
    return p;
  };
  float*     mean    = (float*)    take(B_*C_*sizeof(float));
  float*     invmax  = (float*)    take(B_*sizeof(float));
  float2*    h1      = (float2*)   take(F_*C_*sizeof(float2));
  float2*    h2      = (float2*)   take(F_*C_*sizeof(float2));
  _Float16*  dftT    = (_Float16*) take((size_t)N1_*K1_*sizeof(_Float16));
  _Float16*  istT    = (_Float16*) take((size_t)N2_*K2_*sizeof(_Float16));
  _Float16*  frames  = (_Float16*) take((size_t)M1_*K1_*sizeof(_Float16));
  float*     spec    = (float*)    take((size_t)M1_*N1_*sizeof(float));
  _Float16*  Sg1     = (_Float16*) take((size_t)M2_*K2_*sizeof(_Float16));
  _Float16*  Sg2     = (_Float16*) take((size_t)M2_*K2_*sizeof(_Float16));
  float*     framesO = (float*)    take((size_t)M2_*N2_*sizeof(float));

  // 1) normalization statistics
  mean_kernel  <<<B_*C_, 256, 0, stream>>>(wav, mean);
  maxabs_kernel<<<B_,    256, 0, stream>>>(wav, mean, invmax);

  // 2) DFT / iDFT matrices + steering vectors
  {
    int tot = N1_*K1_ + N2_*K2_ + 2*F_*C_;
    precompute_kernel<<<(tot + 255)/256, 256, 0, stream>>>(window, dftT, istT, h1, h2);
  }

  // 3) framing -> f16 A matrix
  frame_kernel<<<(M1_*K1_ + 255)/256, 256, 0, stream>>>(wav, mean, invmax, frames);

  // 4) STFT GEMM (WMMA): spec[M1,N1] = frames[M1,K1] x dftT^T
  {
    int waves = (M1_/16) * (N1_/64);
    dim3 blk(32, 8);
    wmma_gemm_kernel<<<(waves + 7)/8, blk, 0, stream>>>(frames, dftT, spec, M1_, N1_, K1_);
  }

  // 5) zero padded spectra, then MVDR scan (fills valid entries)
  zero_half_kernel<<<((M2_*K2_) + 255)/256, 256, 0, stream>>>(Sg1, M2_*K2_);
  zero_half_kernel<<<((M2_*K2_) + 255)/256, 256, 0, stream>>>(Sg2, M2_*K2_);
  mvdr_scan_kernel<<<(B_*F_ + 255)/256, 256, 0, stream>>>(spec, h1, h2, Sg1, Sg2);

  // 6) ISTFT GEMM (WMMA) + overlap-add, per beam
  {
    int waves = (M2_/16) * (N2_/64);
    dim3 blk(32, 8);
    wmma_gemm_kernel<<<(waves + 7)/8, blk, 0, stream>>>(Sg1, istT, framesO, M2_, N2_, K2_);
    ola_kernel<<<(OUTLEN_ + 255)/256, 256, 0, stream>>>(framesO, out);
    wmma_gemm_kernel<<<(waves + 7)/8, blk, 0, stream>>>(Sg2, istT, framesO, M2_, N2_, K2_);
    ola_kernel<<<(OUTLEN_ + 255)/256, 256, 0, stream>>>(framesO, out + OUTLEN_);
  }
}
